// MambaModel_62345745269154
// MI455X (gfx1250) — matrix operand order repaired and checked by reading, not compiled, not run on hardware
//
#include <hip/hip_runtime.h>
#include <hip/hip_bf16.h>

typedef __attribute__((ext_vector_type(16))) _Float16 v16h;
typedef __attribute__((ext_vector_type(8)))  _Float16 v8h;
typedef __attribute__((ext_vector_type(8)))  float    v8f;
typedef __attribute__((ext_vector_type(4)))  int      v4i;

#define BATCH 8
#define SEQ   1024
#define BL    (BATCH * SEQ)      // 8192 tokens
#define DM    1024               // d_model
#define DI    2048               // d_inner
#define DS    16                 // d_state
#define DR    16                 // dt_rank
#define DCONV 4
#define XPAD  64                 // padded dt_rank + 2*d_state (48 -> 64)
#define HO    768                // HORIZON*OUT_FEAT
#define LN_EPS 1e-5f

#define AS1 __attribute__((address_space(1)))
#define AS3 __attribute__((address_space(3)))

#if defined(__has_builtin)
#if __has_builtin(__builtin_amdgcn_global_load_async_to_lds_b128)
#define HAVE_ASYNC_LDS 1
#endif
#endif

__device__ __forceinline__ float silu_(float x) { return x / (1.0f + __expf(-x)); }

// ---------------------------------------------------------------------------
// elementwise conversions / fills
// ---------------------------------------------------------------------------
__global__ void k_f32_to_f16(const float* __restrict__ in, _Float16* __restrict__ out, int n) {
    int i = blockIdx.x * blockDim.x + threadIdx.x;
    if (i < n) out[i] = (_Float16)in[i];
}

__global__ void k_cvt_pad(const float* __restrict__ in, _Float16* __restrict__ out,
                          int R, int C, int Cpad) {
    int i = blockIdx.x * blockDim.x + threadIdx.x;
    if (i >= R * Cpad) return;
    int r = i / Cpad, c = i % Cpad;
    out[i] = (c < C) ? (_Float16)in[(size_t)r * C + c] : (_Float16)0.0f;
}

__global__ void k_fill_f16(_Float16* __restrict__ p, int n) {
    int i = blockIdx.x * blockDim.x + threadIdx.x;
    if (i < n) p[i] = (_Float16)0.0f;
}

__global__ void k_copy_pad_f32(const float* __restrict__ in, float* __restrict__ out,
                               int n, int npad) {
    int i = blockIdx.x * blockDim.x + threadIdx.x;
    if (i < npad) out[i] = (i < n) ? in[i] : 0.0f;
}

// transpose + convert + zero-pad: out[c][r] = in[r][c] (guards pad with zeros)
__global__ __launch_bounds__(256)
void k_transpose_cvt(const float* __restrict__ in, _Float16* __restrict__ out,
                     int R, int C, int Rpad, int Cpad) {
    __shared__ float tile[32][33];
    const int c0 = blockIdx.x * 32, r0 = blockIdx.y * 32;
    const int tx = threadIdx.x, ty = threadIdx.y;
#pragma unroll
    for (int i = 0; i < 4; ++i) {
        int r = r0 + ty + i * 8, c = c0 + tx;
        tile[ty + i * 8][tx] = (r < R && c < C) ? in[(size_t)r * C + c] : 0.0f;
    }
    __syncthreads();
#pragma unroll
    for (int i = 0; i < 4; ++i) {
        int c = c0 + ty + i * 8, r = r0 + tx;
        if (c < Cpad && r < Rpad)
            out[(size_t)c * Rpad + r] = (_Float16)tile[tx][ty + i * 8];
    }
}

// ---------------------------------------------------------------------------
// WMMA GEMM:  C[M,N] = A[M,K] * Bt[N,K]^T + bias,  fp16 in / fp32 out
// A buffer has gridDim.y*128 rows; N%64==0; K%64==0; lda/ldb%8==0; 16B aligned.
// 256 threads = 8 waves; macro-tile 128x64; wave: 32x32 (2x2 WMMA); K-step 64.
// Double-buffered LDS; async global->LDS DMA when available, else reg-staged.
// ---------------------------------------------------------------------------
__global__ __launch_bounds__(256)
void k_gemm_wmma(const _Float16* __restrict__ A, int lda,
                 const _Float16* __restrict__ Bt, int ldb,
                 const float* __restrict__ bias,
                 float* __restrict__ C, int ldc,
                 int Mstore, int K, int act)
{
    // row stride 72 halves = 144B: 16B-aligned, fragment reads bank-conflict-free
    __shared__ _Float16 As[2][128][72];
    __shared__ _Float16 Bs[2][64][72];

    const int tid  = threadIdx.x;
    const int lane = tid & 31;          // wave32
    const int wid  = tid >> 5;          // 0..7
    const int wm   = wid >> 1;          // 0..3 : 32-row strip
    const int wn   = wid & 1;           // 0..1 : 32-col strip
    const int rowBase = blockIdx.y * 128;
    const int colBase = blockIdx.x * 64;

    const int hi  = (lane >> 4) & 1;
    const int l15 = lane & 15;
    const int akb = hi * 8;             // A frag K-sub-base per ISA 7.12.2
    const int bkb = hi * 8;             // B frag K-sub-base

    // per-thread staging coords: one v8h (16B) chunk; 8 lanes cover a 128B row
    const int ar = tid >> 3;            // 0..31
    const int kg = (tid & 7) << 3;      // 0,8,...,56
    const _Float16* aP = A  + (size_t)(rowBase + ar) * lda + kg;
    const _Float16* bP = Bt + (size_t)(colBase + ar) * ldb + kg;

    v8f acc[2][2];
#pragma unroll
    for (int i = 0; i < 2; ++i)
#pragma unroll
        for (int j = 0; j < 2; ++j)
#pragma unroll
            for (int e = 0; e < 8; ++e) acc[i][j][e] = 0.0f;

    // ---- 8 WMMAs on LDS buffer `buf` ----
    auto compute = [&](int buf) {
#pragma unroll
        for (int kh = 0; kh < 2; ++kh) {
            const int kb = kh * 32;
            v16h af[2], bf[2];
#pragma unroll
            for (int s = 0; s < 2; ++s) {
                const int arow = wm * 32 + s * 16 + l15;
                v8h alo = *(const v8h*)&As[buf][arow][kb + akb];
                v8h ahi = *(const v8h*)&As[buf][arow][kb + 16 + akb];
                af[s] = __builtin_shufflevector(alo, ahi,
                        0,1,2,3,4,5,6,7,8,9,10,11,12,13,14,15);
                const int bcol = wn * 32 + s * 16 + l15;
                v8h blo = *(const v8h*)&Bs[buf][bcol][kb + bkb];
                v8h bhi = *(const v8h*)&Bs[buf][bcol][kb + 16 + bkb];
                bf[s] = __builtin_shufflevector(blo, bhi,
                        0,1,2,3,4,5,6,7,8,9,10,11,12,13,14,15);
            }
#pragma unroll
            for (int i = 0; i < 2; ++i)
#pragma unroll
                for (int j = 0; j < 2; ++j)
                    acc[i][j] = __builtin_amdgcn_wmma_f32_16x16x32_f16(
                        false, af[i], false, bf[j],
                        (short)0, acc[i][j], false, false);
        }
    };

#if HAVE_ASYNC_LDS
    // ---- CDNA5 async DMA path: global -> LDS with no VGPR roundtrip ----
    auto issue = [&](int koff, int buf) {
#pragma unroll
        for (int s = 0; s < 4; ++s)
            __builtin_amdgcn_global_load_async_to_lds_b128(
                (AS1 v4i*)(aP + koff + (size_t)s * 32 * lda),
                (AS3 v4i*)&As[buf][ar + s * 32][kg], 0, 0);
#pragma unroll
        for (int s = 0; s < 2; ++s)
            __builtin_amdgcn_global_load_async_to_lds_b128(
                (AS1 v4i*)(bP + koff + (size_t)s * 32 * ldb),
                (AS3 v4i*)&Bs[buf][ar + s * 32][kg], 0, 0);
    };
    issue(0, 0);
    int cur = 0;
    for (int k0 = 0; k0 < K; k0 += 64, cur ^= 1) {
#if __has_builtin(__builtin_amdgcn_s_wait_asynccnt)
        __builtin_amdgcn_s_wait_asynccnt(0);
#else
        asm volatile("s_wait_asynccnt 0x0" ::: "memory");
#endif
        __syncthreads();                       // tile `cur` ready; prev reads done
        if (k0 + 64 < K) issue(k0 + 64, cur ^ 1);  // DMA next tile under compute
        compute(cur);
    }
#else
    // ---- fallback: register-staged software pipeline ----
    v8h ra[4], rb[2];
#pragma unroll
    for (int s = 0; s < 4; ++s) ra[s] = *(const v8h*)(aP + (size_t)s * 32 * lda);
#pragma unroll
    for (int s = 0; s < 2; ++s) rb[s] = *(const v8h*)(bP + (size_t)s * 32 * ldb);
#pragma unroll
    for (int s = 0; s < 4; ++s) *(v8h*)&As[0][ar + s * 32][kg] = ra[s];
#pragma unroll
    for (int s = 0; s < 2; ++s) *(v8h*)&Bs[0][ar + s * 32][kg] = rb[s];
    int cur = 0;
    for (int k0 = 0; k0 < K; k0 += 64, cur ^= 1) {
        __syncthreads();                       // tile `cur` visible to all waves
        const bool more = (k0 + 64 < K);
        if (more) {                            // issue next tile's loads (no wait)
#pragma unroll
            for (int s = 0; s < 4; ++s)
                ra[s] = *(const v8h*)(aP + k0 + 64 + (size_t)s * 32 * lda);
#pragma unroll
            for (int s = 0; s < 2; ++s)
                rb[s] = *(const v8h*)(bP + k0 + 64 + (size_t)s * 32 * ldb);
        }
        compute(cur);                          // WMMAs overlap the global loads
        if (more) {                            // single loadcnt wait lands here
#pragma unroll
            for (int s = 0; s < 4; ++s) *(v8h*)&As[cur ^ 1][ar + s * 32][kg] = ra[s];
#pragma unroll
            for (int s = 0; s < 2; ++s) *(v8h*)&Bs[cur ^ 1][ar + s * 32][kg] = rb[s];
        }
    }
#endif

    // ---- store: C/D layout VGPR e -> row (hi?8:0)+e, col = lane&15 ----
#pragma unroll
    for (int i = 0; i < 2; ++i) {
        const int rtile = rowBase + wm * 32 + i * 16 + hi * 8;
#pragma unroll
        for (int j = 0; j < 2; ++j) {
            const int col = colBase + wn * 32 + j * 16 + l15;
            const float bv = bias[col];
#pragma unroll
            for (int e = 0; e < 8; ++e) {
                const int row = rtile + e;
                if (row < Mstore) {
                    float v = acc[i][j][e] + bv;
                    if (act == 1) v = (v > 20.0f) ? v : __logf(1.0f + __expf(v));
                    C[(size_t)row * ldc + col] = v;
                }
            }
        }
    }
}

// ---------------------------------------------------------------------------
// depthwise causal conv1d (width 4) + SiLU.  xi = xr[:, 0:DI] with ld=2*DI.
// ---------------------------------------------------------------------------
__global__ void k_conv_silu(const float* __restrict__ xr,
                            const float* __restrict__ cw,
                            const float* __restrict__ cb,
                            float* __restrict__ xa)
{
    size_t i = (size_t)blockIdx.x * blockDim.x + threadIdx.x;
    if (i >= (size_t)BL * DI) return;
    const int    d = (int)(i % DI);
    const size_t m = i / DI;
    const int    l = (int)(m % SEQ);
    float acc = cb[d];
#pragma unroll
    for (int j = 0; j < DCONV; ++j) {
        const int back = DCONV - 1 - j;
        if (l - back >= 0)
            acc += xr[(m - back) * (size_t)(2 * DI) + d] * cw[d * DCONV + j];
    }
    xa[i] = silu_(acc);
}

// ---------------------------------------------------------------------------
// selective scan: one thread per (b, d); 16 states in registers; ys aliases delta.
// ---------------------------------------------------------------------------
__global__ __launch_bounds__(256)
void k_scan(const float* __restrict__ delta,
            const float* __restrict__ xdbl,
            const float* __restrict__ xa,
            const float* __restrict__ A_log,
            float* ys)
{
    const int d = blockIdx.x * blockDim.x + threadIdx.x;
    const int b = blockIdx.y;
    if (d >= DI) return;

    float A[DS], h[DS];
#pragma unroll
    for (int n = 0; n < DS; ++n) {
        A[n] = -__expf(A_log[d * DS + n]);
        h[n] = 0.0f;
    }
    for (int t = 0; t < SEQ; ++t) {
        const size_t m  = (size_t)b * SEQ + t;
        const float  dt = delta[m * DI + d];
        const float  xt = xa[m * DI + d];
        const float  dx = dt * xt;
        const float* Bt = xdbl + m * XPAD + DR;
        const float* Ct = Bt + DS;
        float yv = 0.0f;
#pragma unroll
        for (int n = 0; n < DS; ++n) {
            const float dA = __expf(dt * A[n]);
            h[n] = dA * h[n] + dx * Bt[n];
            yv  += h[n] * Ct[n];
        }
        ys[m * DI + d] = yv;
    }
}

// ---------------------------------------------------------------------------
// y16 = f16( (ys + xa*D) * silu(res) ), res = xr[:, DI:2*DI]
// ---------------------------------------------------------------------------
__global__ void k_gate(const float* __restrict__ ys, const float* __restrict__ xa,
                       const float* __restrict__ xr, const float* __restrict__ Dv,
                       _Float16* __restrict__ y16)
{
    size_t i = (size_t)blockIdx.x * blockDim.x + threadIdx.x;
    if (i >= (size_t)BL * DI) return;
    const int    d = (int)(i % DI);
    const size_t m = i / DI;
    float y = ys[i] + xa[i] * Dv[d];
    const float r = xr[m * (size_t)(2 * DI) + DI + d];
    y16[i] = (_Float16)(y * silu_(r));
}

// ---------------------------------------------------------------------------
// LayerNorm of the LAST token per batch only -> f16 rows 0..7 of padded buffer
// ---------------------------------------------------------------------------
__global__ __launch_bounds__(256)
void k_ln_last(const float* __restrict__ h, const float* __restrict__ g,
               const float* __restrict__ bparm, _Float16* __restrict__ out)
{
    const int bidx = blockIdx.x;
    const float* row = h + ((size_t)bidx * SEQ + (SEQ - 1)) * DM;
    __shared__ float red[256];
    const int tid = threadIdx.x;

    float s = 0.0f;
    for (int i = tid; i < DM; i += 256) s += row[i];
    red[tid] = s; __syncthreads();
    for (int st = 128; st > 0; st >>= 1) { if (tid < st) red[tid] += red[tid + st]; __syncthreads(); }
    const float mu = red[0] / DM;
    __syncthreads();

    float v = 0.0f;
    for (int i = tid; i < DM; i += 256) { float dd = row[i] - mu; v += dd * dd; }
    red[tid] = v; __syncthreads();
    for (int st = 128; st > 0; st >>= 1) { if (tid < st) red[tid] += red[tid + st]; __syncthreads(); }
    const float rstd = rsqrtf(red[0] / DM + LN_EPS);

    for (int i = tid; i < DM; i += 256)
        out[(size_t)bidx * DM + i] = (_Float16)((row[i] - mu) * rstd * g[i] + bparm[i]);
}

// ---------------------------------------------------------------------------
// host: pipeline
// ---------------------------------------------------------------------------
static inline void cvt16(const float* src, _Float16* dst, int n, hipStream_t s) {
    k_f32_to_f16<<<(n + 255) / 256, 256, 0, s>>>(src, dst, n);
}
static inline void transpose_cvt(const float* in, _Float16* out, int R, int C,
                                 int Rpad, int Cpad, hipStream_t s) {
    dim3 g(Cpad / 32, Rpad / 32), b(32, 8);
    k_transpose_cvt<<<g, b, 0, s>>>(in, out, R, C, Rpad, Cpad);
}
static inline void gemm(const _Float16* A, int lda, const _Float16* Bt, int ldb,
                        const float* bias, float* C, int ldc,
                        int Mstore, int Mpad, int N, int K, int act, hipStream_t s) {
    dim3 g(N / 64, Mpad / 128);
    k_gemm_wmma<<<g, 256, 0, s>>>(A, lda, Bt, ldb, bias, C, ldc, Mstore, K, act);
}

extern "C" void kernel_launch(void* const* d_in, const int* in_sizes, int n_in,
                              void* d_out, int out_size, void* d_ws, size_t ws_size,
                              hipStream_t stream) {
    (void)in_sizes; (void)n_in; (void)out_size; (void)ws_size;
    const float* x     = (const float*)d_in[0];
    const float* emb_w = (const float*)d_in[1];
    const float* emb_b = (const float*)d_in[2];
    const float* blk[2][12];
    for (int b = 0; b < 2; ++b)
        for (int j = 0; j < 12; ++j) blk[b][j] = (const float*)d_in[3 + b * 12 + j];
    const float* ln_g   = (const float*)d_in[27];
    const float* ln_b   = (const float*)d_in[28];
    const float* head_w = (const float*)d_in[29];
    const float* head_b = (const float*)d_in[30];

    // ---- workspace layout (256B aligned) ----
    char* ws = (char*)d_ws;
    size_t off = 0;
    auto alloc = [&](size_t bytes) -> void* {
        off = (off + 255) & ~(size_t)255;
        void* p = ws + off;
        off += bytes;
        return p;
    };
    float*    h_buf    = (float*)   alloc((size_t)BL * DM * 4);
    _Float16* h16      = (_Float16*)alloc((size_t)BL * DM * 2);
    float*    xr       = (float*)   alloc((size_t)BL * 2 * DI * 4);
    float*    xa       = (float*)   alloc((size_t)BL * DI * 4);
    _Float16* xa16     = (_Float16*)alloc((size_t)BL * DI * 2);   // reused as y16
    float*    delta    = (float*)   alloc((size_t)BL * DI * 4);   // reused as ys (in-place)
    float*    xdbl     = (float*)   alloc((size_t)BL * XPAD * 4);
    _Float16* xdbl16   = (_Float16*)alloc((size_t)BL * XPAD * 2);
    _Float16* x16      = (_Float16*)alloc((size_t)BL * 64 * 2);   // K padded 32->64
    _Float16* ln16     = (_Float16*)alloc((size_t)128 * DM * 2);  // M padded 8->128
    float*    xp_b_pad = (float*)   alloc((size_t)XPAD * 4);
    _Float16* emb_wt   = (_Float16*)alloc((size_t)DM * 64 * 2);   // [1024, 64]
    _Float16* head_wt  = (_Float16*)alloc((size_t)HO * DM * 2);   // [768, 1024]
    _Float16* in_wt[2], *xp_wt[2], *dt_wt[2], *out_wt[2];
    for (int b = 0; b < 2; ++b) {
        in_wt[b]  = (_Float16*)alloc((size_t)(2 * DI) * DM * 2);  // [4096, 1024]
        xp_wt[b]  = (_Float16*)alloc((size_t)XPAD * DI * 2);      // [64, 2048]
        dt_wt[b]  = (_Float16*)alloc((size_t)DI * 64 * 2);        // [2048, 64]
        out_wt[b] = (_Float16*)alloc((size_t)DM * DI * 2);        // [1024, 2048]
    }

    // ---- weight transposition + conversion (per launch; deterministic) ----
    transpose_cvt(emb_w,  emb_wt,  32,  DM,     64,  DM,     stream);
    transpose_cvt(head_w, head_wt, DM,  HO,     DM,  HO,     stream);
    for (int b = 0; b < 2; ++b) {
        transpose_cvt(blk[b][0],  in_wt[b],  DM, 2 * DI, DM, 2 * DI, stream);
        transpose_cvt(blk[b][4],  xp_wt[b],  DI, 48,     DI, XPAD,   stream);
        transpose_cvt(blk[b][6],  dt_wt[b],  DR, DI,     64, DI,     stream);
        transpose_cvt(blk[b][10], out_wt[b], DI, DM,     DI, DM,     stream);
    }

    // ---- embedding: h = x @ emb_w + emb_b (K padded to 64) ----
    k_cvt_pad<<<(BL * 64 + 255) / 256, 256, 0, stream>>>(x, x16, BL, 32, 64);
    gemm(x16, 64, emb_wt, 64, emb_b, h_buf, DM, BL, BL, DM, 64, 0, stream);

    const int ew = (BL * DI + 255) / 256;

    // ---- two mamba blocks ----
    for (int b = 0; b < 2; ++b) {
        const float* in_b_  = blk[b][1];
        const float* conv_w = blk[b][2];
        const float* conv_b = blk[b][3];
        const float* xp_b   = blk[b][5];
        const float* dt_b   = blk[b][7];
        const float* A_log  = blk[b][8];
        const float* Dv     = blk[b][9];
        const float* out_b_ = blk[b][11];

        k_copy_pad_f32<<<1, 64, 0, stream>>>(xp_b, xp_b_pad, 48, XPAD);

        cvt16(h_buf, h16, BL * DM, stream);
        // in_proj -> xr [BL, 4096] (xi | res)
        gemm(h16, DM, in_wt[b], DM, in_b_, xr, 2 * DI, BL, BL, 2 * DI, DM, 0, stream);
        // depthwise causal conv + silu -> xa
        k_conv_silu<<<ew, 256, 0, stream>>>(xr, conv_w, conv_b, xa);
        cvt16(xa, xa16, BL * DI, stream);
        // x_proj -> xdbl [BL, 64] (cols 48..63 land zero via padded weights/bias)
        gemm(xa16, DI, xp_wt[b], DI, xp_b_pad, xdbl, XPAD, BL, BL, XPAD, DI, 0, stream);
        cvt16(xdbl, xdbl16, BL * XPAD, stream);
        // dt_proj + softplus -> delta (K padded to 64, padded weight cols are zero)
        gemm(xdbl16, XPAD, dt_wt[b], 64, dt_b, delta, DI, BL, BL, DI, 64, 1, stream);
        // selective scan (ys written in-place over delta)
        k_scan<<<dim3(DI / 256, BATCH), 256, 0, stream>>>(delta, xdbl, xa, A_log, delta);
        // gate -> y16 (aliases xa16)
        k_gate<<<ew, 256, 0, stream>>>(delta, xa, xr, Dv, xa16);
        // out_proj -> h
        gemm(xa16, DI, out_wt[b], DI, out_b_, h_buf, DM, BL, BL, DM, DI, 0, stream);
    }

    // ---- LN of last token + head (A padded to 128 zero rows) ----
    k_fill_f16<<<(128 * DM + 255) / 256, 256, 0, stream>>>(ln16, 128 * DM);
    k_ln_last<<<BATCH, 256, 0, stream>>>(h_buf, ln_g, ln_b, ln16);
    gemm(ln16, DM, head_wt, DM, head_b, (float*)d_out, HO, BATCH, 128, HO, DM, 0, stream);
}
